// LogicGatedSNN_47545287967003
// MI455X (gfx1250) — matrix operand order repaired and checked by reading, not compiled, not run on hardware
//
#include <hip/hip_runtime.h>

typedef __attribute__((ext_vector_type(16))) _Float16 v16h;
typedef __attribute__((ext_vector_type(8)))  float    v8f;
typedef __attribute__((ext_vector_type(4)))  float    v4f;

#define IN_F   8192
#define OUT_F  8192
#define WTHRESH 50.0f

// 256 threads = 8 wave32; 16 output rows per block; grid = OUT_F/16 = 512 blocks.
__global__ __launch_bounds__(256)
void snn_fused_kernel(const float* __restrict__ x_in,      // [IN_F] spike_input (0/1)
                      const float* __restrict__ syn,       // [OUT_F, IN_F] synapse_states
                      const float* __restrict__ mem,       // [OUT_F] membrane_potential
                      const float* __restrict__ thr,       // [OUT_F] adaptive_threshold
                      const float* __restrict__ trace,     // [OUT_F, IN_F] eligibility_trace
                      float* __restrict__ out_spike,       // [OUT_F]
                      float* __restrict__ out_mem,         // [OUT_F]
                      float* __restrict__ out_thr,         // [OUT_F]
                      float* __restrict__ out_trace)       // [OUT_F, IN_F]
{
    __shared__ float xs[IN_F];          // staged spike_input (32 KB of the 320 KB/WGP)
    __shared__ float partial[8 * 16];   // per-wave partial row sums
    __shared__ float ls_spike[16];      // spikes for this block's rows

    const int tid  = threadIdx.x;
    const int wave = tid >> 5;
    const int lane = tid & 31;          // wave32
    const int l    = lane & 15;
    const int hi   = lane >> 4;
    const int row_base = blockIdx.x * 16;

    // ---- stage x into LDS (B128 loads) ----
    for (int i = tid * 4; i < IN_F; i += 256 * 4) {
        *(v4f*)&xs[i] = *(const v4f*)&x_in[i];
    }
    __syncthreads();

    // ---- binary GEMV via v_wmma_f32_16x16x32_f16 ----
    // Each wave reduces K slice [wave*1024, wave*1024+1024) for rows row_base..row_base+15.
    v8f c = {};
    const int    row  = row_base + l;
    const float* wrow = syn + (size_t)row * IN_F;
    const int kbeg = wave * 1024;
    const int kend = kbeg + 1024;

    for (int k0 = kbeg; k0 < kend; k0 += 32) {
        // A (16-bit 16x32 layout): halves 0..7 -> K = k0 + hi*8 + j
        //                          halves 8..15 -> K = k0 + 16 + hi*8 + j
        v4f a0 = __builtin_nontemporal_load((const v4f*)(wrow + k0 + hi * 8));
        v4f a1 = __builtin_nontemporal_load((const v4f*)(wrow + k0 + hi * 8 + 4));
        v4f a2 = __builtin_nontemporal_load((const v4f*)(wrow + k0 + 16 + hi * 8));
        v4f a3 = __builtin_nontemporal_load((const v4f*)(wrow + k0 + 16 + hi * 8 + 4));
        v16h A;
        #pragma unroll
        for (int j = 0; j < 4; ++j) {
            A[j]      = (_Float16)((a0[j] > WTHRESH) ? 1.0f : 0.0f);
            A[4 + j]  = (_Float16)((a1[j] > WTHRESH) ? 1.0f : 0.0f);
            A[8 + j]  = (_Float16)((a2[j] > WTHRESH) ? 1.0f : 0.0f);
            A[12 + j] = (_Float16)((a3[j] > WTHRESH) ? 1.0f : 0.0f);
        }
        // B (32x16), x broadcast to all 16 columns: halves 0..7 -> K = k0 + hi*8 + j, halves 8..15 -> K = k0 + 16 + hi*8 + j
        const float* xb = &xs[k0 + hi * 8];
        v4f b0 = *(const v4f*)(xb);
        v4f b1 = *(const v4f*)(xb + 4);
        v4f b2 = *(const v4f*)(xb + 16);
        v4f b3 = *(const v4f*)(xb + 20);
        v16h B;
        #pragma unroll
        for (int j = 0; j < 4; ++j) {
            B[j]      = (_Float16)b0[j];
            B[4 + j]  = (_Float16)b1[j];
            B[8 + j]  = (_Float16)b2[j];
            B[12 + j] = (_Float16)b3[j];
        }
        c = __builtin_amdgcn_wmma_f32_16x16x32_f16(
                /*neg_a=*/false, A, /*neg_b=*/false, B,
                /*c_mod=*/(short)0, c, /*reuse_a=*/false, /*reuse_b=*/false);
    }

    // All 16 D columns are identical (B columns identical).
    // C/D layout: lanes 0-15 hold M=0..7 in c[0..7]; lanes 16-31 hold M=8..15.
    if (lane == 0) {
        #pragma unroll
        for (int r = 0; r < 8; ++r) partial[wave * 16 + r] = c[r];
    } else if (lane == 16) {
        #pragma unroll
        for (int r = 0; r < 8; ++r) partial[wave * 16 + 8 + r] = c[r];
    }
    __syncthreads();

    // ---- LIF update for this block's 16 rows ----
    if (tid < 16) {
        float cur = 0.0f;
        #pragma unroll
        for (int w = 0; w < 8; ++w) cur += partial[w * 16 + tid];
        const int r = row_base + tid;
        const float v = mem[r] * 0.9f + cur;
        const float s = (v >= thr[r]) ? 1.0f : 0.0f;
        out_spike[r] = s;
        out_mem[r]   = v * (1.0f - s) * 0.1f;
        float nt = thr[r] + (s - 0.1f) * 0.1f;
        out_thr[r]   = fminf(fmaxf(nt, 1.0f), 20.0f);
        ls_spike[tid] = s;
    }
    __syncthreads();

    // ---- eligibility trace: clip(trace*0.8 + s*x, 0, 5), fully streaming (NT) ----
    for (int r = 0; r < 16; ++r) {
        const float  s    = ls_spike[r];
        const size_t base = (size_t)(row_base + r) * IN_F;
        for (int i = tid * 4; i < IN_F; i += 256 * 4) {
            v4f t  = __builtin_nontemporal_load((const v4f*)(trace + base + i));
            v4f xv = *(const v4f*)&xs[i];
            v4f o;
            #pragma unroll
            for (int j = 0; j < 4; ++j) {
                float val = t[j] * 0.8f + s * xv[j];
                o[j] = fminf(fmaxf(val, 0.0f), 5.0f);
            }
            __builtin_nontemporal_store(o, (v4f*)(out_trace + base + i));
        }
    }
}

extern "C" void kernel_launch(void* const* d_in, const int* in_sizes, int n_in,
                              void* d_out, int out_size, void* d_ws, size_t ws_size,
                              hipStream_t stream) {
    (void)in_sizes; (void)n_in; (void)out_size; (void)d_ws; (void)ws_size;
    const float* x_in  = (const float*)d_in[0];  // spike_input        [8192]
    const float* syn   = (const float*)d_in[1];  // synapse_states     [8192*8192]
    const float* mem   = (const float*)d_in[2];  // membrane_potential [8192]
    const float* thr   = (const float*)d_in[3];  // adaptive_threshold [8192]
    const float* trace = (const float*)d_in[4];  // eligibility_trace  [8192*8192]

    float* out        = (float*)d_out;           // spikes, new_mem, new_thr, new_trace (flat)
    float* out_spike  = out;
    float* out_mem    = out + OUT_F;
    float* out_thr    = out + 2 * OUT_F;
    float* out_trace  = out + 3 * OUT_F;

    snn_fused_kernel<<<OUT_F / 16, 256, 0, stream>>>(
        x_in, syn, mem, thr, trace, out_spike, out_mem, out_thr, out_trace);
}
